// StateBayesianKalmanNet_89472758710314
// MI455X (gfx1250) — matrix operand order repaired and checked by reading, not compiled, not run on hardware
//
#include <hip/hip_runtime.h>
#include <hip/hip_bf16.h>
#include <math.h>

#define SS   10
#define BB   256
#define SD   16
#define OD   8
#define HH   3200
#define H3   9600
#define OUT1 512
#define INW  48
#define TEMP_INV 10.0f   // 1 / 0.1

typedef __attribute__((ext_vector_type(16))) __bf16 bf16x16;
typedef __attribute__((ext_vector_type(8)))  __bf16 bf16x8;
typedef __attribute__((ext_vector_type(8)))  float  f32x8;

union FragU { bf16x16 v; bf16x8 h[2]; };

__device__ __forceinline__ float sigmoidf(float x) { return 1.0f / (1.0f + expf(-x)); }

// A fragment: 16x32 bf16 (MxK). p0 -> element (m0, k0), row stride ld.
// ISA 7.12.2: lanes 0-15 hold row m=lane, k {0..7}+{16..23}; lanes 16-31 hold k {8..15}+{24..31}.
__device__ __forceinline__ bf16x16 load_a_frag(const __bf16* p0, int ld, int lane) {
  const int m  = lane & 15;
  const int hi = lane >> 4;
  const __bf16* p = p0 + m * ld + hi * 8;
  FragU f;
  f.h[0] = *(const bf16x8*)(p);
  f.h[1] = *(const bf16x8*)(p + 16);
  return f.v;
}

// B fragment: 32x16 bf16 (KxN) built from row-major W (rows = N, K contiguous).
// w0 -> element (row n0, col k0). Lane holds col n=lane&15, k {0..7}+{16..23} (lanes 0-15) / {8..15}+{24..31} (lanes 16-31).
__device__ __forceinline__ bf16x16 load_b_frag(const __bf16* w0, int ld, int lane) {
  const int n  = lane & 15;
  const int hi = lane >> 4;
  const __bf16* p = w0 + n * ld + hi * 8;
  FragU f;
  f.h[0] = *(const bf16x8*)(p);
  f.h[1] = *(const bf16x8*)(p + 16);
  return f.v;
}

__device__ __forceinline__ f32x8 wmma_bf16(bf16x16 a, bf16x16 b, f32x8 c) {
  return __builtin_amdgcn_wmma_f32_16x16x32_bf16(false, a, false, b, (short)0, c, false, false);
}

// ---------------- prologue: x_predicted, residual, inp (B x 48) ----------------
__global__ void prologue_kernel(const float* __restrict__ y_t, const float* __restrict__ xfp,
                                const float* __restrict__ xfpp, const float* __restrict__ yprev,
                                const float* __restrict__ dxp, const float* __restrict__ Fm,
                                const float* __restrict__ Hm, float* __restrict__ xpred,
                                float* __restrict__ resid, float* __restrict__ inp) {
  const int b = threadIdx.x;           // 256 threads, one per batch row
  float xp[SD];
  for (int i = 0; i < SD; ++i) {
    float t = 0.f;
    for (int k = 0; k < SD; ++k) t += xfp[b * SD + k] * Fm[i * SD + k];
    xp[i] = tanhf(t);
    xpred[b * SD + i] = xp[i];
    inp[b * INW + i]      = dxp[b * SD + i];
    inp[b * INW + 24 + i] = xfp[b * SD + i] - xfpp[b * SD + i];
  }
  for (int o = 0; o < OD; ++o) {
    float t = 0.f;
    for (int i = 0; i < SD; ++i) t += xp[i] * Hm[o * SD + i];
    float r = y_t[b * OD + o] - t;
    resid[b * OD + o] = r;
    inp[b * INW + SD + o] = r;
    inp[b * INW + 40 + o] = yprev[b * OD + o];
  }
}

// ---------------- a = relu(inp @ fc_in_w.T + b) ----------------
__global__ void fcin_kernel(const float* __restrict__ inp, const float* __restrict__ w,
                            const float* __restrict__ b, float* __restrict__ a) {
  const int h = blockIdx.x * blockDim.x + threadIdx.x;   // HH
  const int m = blockIdx.y;                              // BB
  float acc = b[h];
  const float* ir = inp + m * INW;
  const float* wr = w + h * INW;
#pragma unroll
  for (int k = 0; k < INW; ++k) acc += ir[k] * wr[k];
  a[m * HH + h] = fmaxf(acc, 0.0f);
}

// ---------------- fp32 -> bf16 weight conversion ----------------
__global__ void cvt_kernel(const float* __restrict__ src, __bf16* __restrict__ dst, int n) {
  int i = blockIdx.x * blockDim.x + threadIdx.x;
  const int stride = gridDim.x * blockDim.x;
  for (; i < n; i += stride) dst[i] = (__bf16)src[i];
}

// ---------------- build X = dropout(a, u1) and Hb = bf16(h_prev) ----------------
__global__ void dropx_kernel(const float* __restrict__ a, const float* __restrict__ u1,
                             const float* __restrict__ hprev, const float* __restrict__ plog1,
                             __bf16* __restrict__ X, __bf16* __restrict__ Hb) {
  const int idx = blockIdx.x * blockDim.x + threadIdx.x;   // SS*BB*HH exactly
  const float l1 = plog1[0];                 // log(p)-log1p(-p) == logit
  const float p1 = sigmoidf(l1);
  const float scale = 1.0f / (1.0f - p1);
  const int r = idx % (BB * HH);
  const float u = u1[idx];
  const float zm = sigmoidf((l1 + logf(u) - log1pf(-u)) * TEMP_INV);
  X[idx]  = (__bf16)(a[r] * (1.0f - zm) * scale);
  Hb[idx] = (__bf16)hprev[idx];
}

// ---------------- GRU: 6 fused WMMA chains + gates (one 16x16 h_new tile / wave) ----------------
__global__ __launch_bounds__(512)
void gru_kernel(const __bf16* __restrict__ Xall, const __bf16* __restrict__ Hall,
                const __bf16* __restrict__ Wih, const __bf16* __restrict__ Whh,
                const float* __restrict__ bih, const float* __restrict__ bhh,
                const float* __restrict__ hprev, __bf16* __restrict__ Hnew) {
  const int s    = blockIdx.y;
  const int j0   = blockIdx.x << 4;
  const int lane = threadIdx.x & 31;
  const int m0   = (threadIdx.x >> 5) << 4;   // 16 waves cover all 256 batch rows

  const __bf16* X  = Xall + (size_t)s * (BB * HH) + (size_t)m0 * HH;
  const __bf16* Hs = Hall + (size_t)s * (BB * HH) + (size_t)m0 * HH;
  const __bf16* Wr = Wih + (size_t)j0 * HH;
  const __bf16* Wz = Wih + (size_t)(HH + j0) * HH;
  const __bf16* Wn = Wih + (size_t)(2 * HH + j0) * HH;
  const __bf16* Vr = Whh + (size_t)j0 * HH;
  const __bf16* Vz = Whh + (size_t)(HH + j0) * HH;
  const __bf16* Vn = Whh + (size_t)(2 * HH + j0) * HH;

  f32x8 axr = {0,0,0,0,0,0,0,0};
  f32x8 axz = axr, axn = axr, ahr = axr, ahz = axr, ahn = axr;

  for (int k = 0; k < HH; k += 32) {
    bf16x16 ax = load_a_frag(X + k, HH, lane);
    bf16x16 ah = load_a_frag(Hs + k, HH, lane);
    axr = wmma_bf16(ax, load_b_frag(Wr + k, HH, lane), axr);
    axz = wmma_bf16(ax, load_b_frag(Wz + k, HH, lane), axz);
    axn = wmma_bf16(ax, load_b_frag(Wn + k, HH, lane), axn);
    ahr = wmma_bf16(ah, load_b_frag(Vr + k, HH, lane), ahr);
    ahz = wmma_bf16(ah, load_b_frag(Vz + k, HH, lane), ahz);
    ahn = wmma_bf16(ah, load_b_frag(Vn + k, HH, lane), ahn);
  }

  const int col = j0 + (lane & 15);
  const float bir = bih[col], biz = bih[col + HH], bin = bih[col + 2 * HH];
  const float bhr = bhh[col], bhz = bhh[col + HH], bhn = bhh[col + 2 * HH];
  const int mbase = m0 + ((lane >> 4) << 3);
#pragma unroll
  for (int i = 0; i < 8; ++i) {
    const int m = mbase + i;
    const float r = sigmoidf(axr[i] + bir + ahr[i] + bhr);
    const float z = sigmoidf(axz[i] + biz + ahz[i] + bhz);
    const float n = tanhf(axn[i] + bin + r * (ahn[i] + bhn));
    const float hp = hprev[((size_t)s * BB + m) * HH + col];
    Hnew[((size_t)s * BB + m) * HH + col] = (__bf16)((1.0f - z) * n + z * hp);
  }
}

// ---------------- fc1 + relu + concrete dropout (bf16 out) ----------------
__global__ __launch_bounds__(512)
void fc1_kernel(const __bf16* __restrict__ Hnew, const __bf16* __restrict__ W1,
                const float* __restrict__ b1, const float* __restrict__ u2,
                const float* __restrict__ plog2, __bf16* __restrict__ O) {
  const int s    = blockIdx.y;
  const int n0   = blockIdx.x << 4;
  const int lane = threadIdx.x & 31;
  const int m0   = (threadIdx.x >> 5) << 4;

  const __bf16* A  = Hnew + (size_t)s * (BB * HH) + (size_t)m0 * HH;
  const __bf16* Bw = W1 + (size_t)n0 * HH;
  f32x8 acc = {0,0,0,0,0,0,0,0};
  for (int k = 0; k < HH; k += 32)
    acc = wmma_bf16(load_a_frag(A + k, HH, lane), load_b_frag(Bw + k, HH, lane), acc);

  const float l2 = plog2[0];
  const float p2 = sigmoidf(l2);
  const float inv1mp2 = 1.0f / (1.0f - p2);
  const int col = n0 + (lane & 15);
  const float bias = b1[col];
  const int mbase = m0 + ((lane >> 4) << 3);
#pragma unroll
  for (int i = 0; i < 8; ++i) {
    const int m = mbase + i;
    float o = fmaxf(acc[i] + bias, 0.0f);
    const float u = u2[((size_t)s * BB + m) * OUT1 + col];
    const float zm = sigmoidf((l2 + logf(u) - log1pf(-u)) * TEMP_INV);
    o *= (1.0f - zm) * inv1mp2;
    O[((size_t)s * BB + m) * OUT1 + col] = (__bf16)o;
  }
}

// ---------------- fc2 -> Kalman gain (fp32) ----------------
__global__ __launch_bounds__(512)
void fc2_kernel(const __bf16* __restrict__ O, const __bf16* __restrict__ W2,
                const float* __restrict__ b2, float* __restrict__ Kmat) {
  const int s    = blockIdx.y;
  const int n0   = blockIdx.x << 4;
  const int lane = threadIdx.x & 31;
  const int m0   = (threadIdx.x >> 5) << 4;

  const __bf16* A  = O + (size_t)s * (BB * OUT1) + (size_t)m0 * OUT1;
  const __bf16* Bw = W2 + (size_t)n0 * OUT1;
  f32x8 acc = {0,0,0,0,0,0,0,0};
  for (int k = 0; k < OUT1; k += 32)
    acc = wmma_bf16(load_a_frag(A + k, OUT1, lane), load_b_frag(Bw + k, OUT1, lane), acc);

  const int col = n0 + (lane & 15);
  const float bias = b2[col];
  const int mbase = m0 + ((lane >> 4) << 3);
#pragma unroll
  for (int i = 0; i < 8; ++i) {
    const int m = mbase + i;
    Kmat[((size_t)s * BB + m) * (SD * OD) + col] = acc[i] + bias;
  }
}

// ---------------- xf = x_predicted + K @ residual ----------------
__global__ void corr_kernel(const float* __restrict__ Kmat, const float* __restrict__ resid,
                            const float* __restrict__ xpred, float* __restrict__ xf) {
  const int idx = blockIdx.x * blockDim.x + threadIdx.x;   // SS*BB*SD
  const int i  = idx & 15;
  const int sm = idx >> 4;          // s*BB + m
  const int m  = sm % BB;
  float t = xpred[m * SD + i];
  const float* Km = Kmat + (size_t)sm * (SD * OD) + i * OD;
  const float* rr = resid + m * OD;
#pragma unroll
  for (int o = 0; o < OD; ++o) t += Km[o] * rr[o];
  xf[idx] = t;
}

// ---------------- mean over ensemble + covariance ----------------
__global__ __launch_bounds__(256)
void final_kernel(const float* __restrict__ xf, float* __restrict__ out) {
  __shared__ float sx[SS][SD];
  __shared__ float smean[SD];
  const int b = blockIdx.x;
  const int t = threadIdx.x;
  if (t < SS * SD) {
    const int s = t / SD, i = t % SD;
    sx[s][i] = xf[((size_t)s * BB + b) * SD + i];
  }
  __syncthreads();
  if (t < SD) {
    float acc = 0.f;
    for (int s = 0; s < SS; ++s) acc += sx[s][t];
    acc *= (1.0f / SS);
    smean[t] = acc;
    out[b * SD + t] = acc;
  }
  __syncthreads();
  const int i = t >> 4, j = t & 15;
  float p = 0.f;
  for (int s = 0; s < SS; ++s) p += (sx[s][i] - smean[i]) * (sx[s][j] - smean[j]);
  out[BB * SD + b * SD * SD + t] = p * (1.0f / SS);
}

// ---------------- deterministic regularizer reduction ----------------
__global__ __launch_bounds__(1024)
void reg_kernel(const float* __restrict__ w_in, const float* __restrict__ w1,
                const float* __restrict__ pl1, const float* __restrict__ pl2,
                float* __restrict__ out) {
  __shared__ float red[1024];
  const int t = threadIdx.x;
  float s1 = 0.f;
  for (int i = t; i < HH * INW; i += 1024) { float v = w_in[i]; s1 += v * v; }
  float s2 = 0.f;
  for (int i = t; i < OUT1 * HH; i += 1024) { float v = w1[i]; s2 += v * v; }
  red[t] = s1; __syncthreads();
  for (int off = 512; off > 0; off >>= 1) { if (t < off) red[t] += red[t + off]; __syncthreads(); }
  const float S1 = red[0]; __syncthreads();
  red[t] = s2; __syncthreads();
  for (int off = 512; off > 0; off >>= 1) { if (t < off) red[t] += red[t + off]; __syncthreads(); }
  if (t == 0) {
    const float S2 = red[0];
    const float p1 = sigmoidf(pl1[0]), p2 = sigmoidf(pl2[0]);
    const float e1 = p1 * logf(p1) + (1.0f - p1) * log1pf(-p1);
    const float e2 = p2 * logf(p2) + (1.0f - p2) * log1pf(-p2);
    const float reg = S1 / (1.0f - p1) + (float)HH * e1 + S2 / (1.0f - p2) + (float)OUT1 * e2;
    for (int s = 0; s < SS; ++s) out[BB * SD + BB * SD * SD + s] = reg;
  }
}

extern "C" void kernel_launch(void* const* d_in, const int* in_sizes, int n_in,
                              void* d_out, int out_size, void* d_ws, size_t ws_size,
                              hipStream_t stream) {
  (void)in_sizes; (void)n_in; (void)out_size; (void)ws_size;
  const float* y_t    = (const float*)d_in[0];
  const float* xfp    = (const float*)d_in[1];
  const float* xfpp   = (const float*)d_in[2];
  const float* yprev  = (const float*)d_in[3];
  const float* dxp    = (const float*)d_in[4];
  const float* hprev  = (const float*)d_in[5];
  const float* Fm     = (const float*)d_in[6];
  const float* Hm     = (const float*)d_in[7];
  const float* fcin_w = (const float*)d_in[8];
  const float* fcin_b = (const float*)d_in[9];
  const float* Wih    = (const float*)d_in[10];
  const float* Whh    = (const float*)d_in[11];
  const float* bih    = (const float*)d_in[12];
  const float* bhh    = (const float*)d_in[13];
  const float* fc1w   = (const float*)d_in[14];
  const float* fc1b   = (const float*)d_in[15];
  const float* fc2w   = (const float*)d_in[16];
  const float* fc2b   = (const float*)d_in[17];
  const float* pl1    = (const float*)d_in[18];
  const float* pl2    = (const float*)d_in[19];
  const float* u1     = (const float*)d_in[20];
  const float* u2     = (const float*)d_in[21];
  float* out = (float*)d_out;

  char* p = (char*)d_ws;
  auto alloc = [&](size_t bytes) { char* r = p; p += (bytes + 255) & ~(size_t)255; return r; };
  __bf16* Wih_b = (__bf16*)alloc((size_t)H3 * HH * 2);
  __bf16* Whh_b = (__bf16*)alloc((size_t)H3 * HH * 2);
  __bf16* W1_b  = (__bf16*)alloc((size_t)OUT1 * HH * 2);
  __bf16* W2_b  = (__bf16*)alloc((size_t)SD * OD * OUT1 * 2);
  float*  a_f   = (float*)alloc((size_t)BB * HH * 4);
  __bf16* X_b   = (__bf16*)alloc((size_t)SS * BB * HH * 2);
  __bf16* H_b   = (__bf16*)alloc((size_t)SS * BB * HH * 2);
  __bf16* Hn_b  = (__bf16*)alloc((size_t)SS * BB * HH * 2);
  __bf16* O_b   = (__bf16*)alloc((size_t)SS * BB * OUT1 * 2);
  float*  Kmat  = (float*)alloc((size_t)SS * BB * SD * OD * 4);
  float*  xf    = (float*)alloc((size_t)SS * BB * SD * 4);
  float*  xpred = (float*)alloc((size_t)BB * SD * 4);
  float*  resid = (float*)alloc((size_t)BB * OD * 4);
  float*  inp   = (float*)alloc((size_t)BB * INW * 4);

  prologue_kernel<<<1, 256, 0, stream>>>(y_t, xfp, xfpp, yprev, dxp, Fm, Hm, xpred, resid, inp);
  fcin_kernel<<<dim3(HH / 64, BB), 64, 0, stream>>>(inp, fcin_w, fcin_b, a_f);
  cvt_kernel<<<2048, 256, 0, stream>>>(Wih, Wih_b, H3 * HH);
  cvt_kernel<<<2048, 256, 0, stream>>>(Whh, Whh_b, H3 * HH);
  cvt_kernel<<<2048, 256, 0, stream>>>(fc1w, W1_b, OUT1 * HH);
  cvt_kernel<<<64, 256, 0, stream>>>(fc2w, W2_b, SD * OD * OUT1);
  dropx_kernel<<<(SS * BB * HH) / 256, 256, 0, stream>>>(a_f, u1, hprev, pl1, X_b, H_b);
  gru_kernel<<<dim3(HH / 16, SS), 512, 0, stream>>>(X_b, H_b, Wih_b, Whh_b, bih, bhh, hprev, Hn_b);
  fc1_kernel<<<dim3(OUT1 / 16, SS), 512, 0, stream>>>(Hn_b, W1_b, fc1b, u2, pl2, O_b);
  fc2_kernel<<<dim3((SD * OD) / 16, SS), 512, 0, stream>>>(O_b, W2_b, fc2b, Kmat);
  corr_kernel<<<(SS * BB * SD) / 256, 256, 0, stream>>>(Kmat, resid, xpred, xf);
  final_kernel<<<BB, 256, 0, stream>>>(xf, out);
  reg_kernel<<<1, 1024, 0, stream>>>(fcin_w, fc1w, pl1, pl2, out);
}
